// TransformerBlock_10333691314606
// MI455X (gfx1250) — matrix operand order repaired and checked by reading, not compiled, not run on hardware
//
#include <hip/hip_runtime.h>
#include <hip/hip_bf16.h>

// ---------------------------------------------------------------------------
// MI455X (gfx1250) transformer block.
// bf16 WMMA (v_wmma_f32_16x16x32_bf16) with f32 accum everywhere.
// GEMM: 4-wave workgroups, 256x64 tiles. B tiles are double-buffered in LDS
// and filled with CDNA5 async copies (global_load_async_to_lds_b128,
// s_wait_asynccnt) so B traffic bypasses VGPRs and is shared by 4 waves.
// A fragments stream from the 192MB L2 as global_load_b128 + prefetch.
// Attention is flash-style with online softmax; P goes C-layout -> LDS ->
// A-layout (ds_store_b16 / ds_load_b128).
// ---------------------------------------------------------------------------

typedef __bf16 bf16;
typedef __attribute__((ext_vector_type(4)))  __bf16 bf16x4;
typedef __attribute__((ext_vector_type(8)))  __bf16 bf16x8;
typedef __attribute__((ext_vector_type(16))) __bf16 bf16x16;
typedef __attribute__((ext_vector_type(8)))  float  f32x8;

#define EMB   1024
#define HEADS 16
#define HD    64
#define FF    4096
#define SEQ   2048
#define BATCH 2
#define TOK   (BATCH*SEQ)   // 4096 tokens

// ----------------------------- WMMA helpers --------------------------------

__device__ __forceinline__ f32x8 fzero() {
    f32x8 z;
#pragma unroll
    for (int i = 0; i < 8; ++i) z[i] = 0.0f;
    return z;
}

__device__ __forceinline__ bf16x16 combine8(bf16x8 lo, bf16x8 hi) {
    bf16x16 f;
#pragma unroll
    for (int i = 0; i < 8; ++i) { f[i] = lo[i]; f[i + 8] = hi[i]; }
    return f;
}

__device__ __forceinline__ f32x8 wmma_bf16(bf16x16 a, bf16x16 b, f32x8 c) {
    // 8 args: (neg_a, A, neg_b, B, c_mod, C, reuse_a, reuse_b)
    return __builtin_amdgcn_wmma_f32_16x16x32_bf16(false, a, false, b,
                                                   (short)0, c, false, false);
}

// A-fragment (16x32, 16-bit): lanes 0-15 row M=lane hold K {0..7, 16..23},
// lanes 16-31 row M=lane-16 hold K {8..15, 24..31}.  base -> element (m0,k0),
// row-major with leading dim ld.  Two 16B loads per lane.
__device__ __forceinline__ bf16x16 load_frag_a(const bf16* base, int ld,
                                               int lane) {
    const int r = lane & 15, h = lane >> 4;
    const bf16* p = base + (size_t)r * ld + h * 8;
    bf16x8 c0 = *(const bf16x8*)(p);
    bf16x8 c1 = *(const bf16x8*)(p + 16);
    return combine8(c0, c1);
}

// B-fragment (32x16, 16-bit) from Bt = B^T stored row-major [N, K]:
// lanes 0-15 col N=lane hold K {0..7, 16..23}; lanes 16-31 hold K {8..15, 24..31}.
// Two 16B loads per lane.
__device__ __forceinline__ bf16x16 load_frag_b(const bf16* base, int ld,
                                               int lane) {
    const int n = lane & 15, h = lane >> 4;
    const bf16* p = base + (size_t)n * ld + h * 8;
    bf16x8 c0 = *(const bf16x8*)(p);
    bf16x8 c1 = *(const bf16x8*)(p + 16);
    return combine8(c0, c1);
}

__device__ __forceinline__ float gelu_tanh(float x) {
    const float c = 0.7978845608028654f;  // sqrt(2/pi)
    float x3 = x * x * x;
    return 0.5f * x * (1.0f + tanhf(c * (x + 0.044715f * x3)));
}

// --------------------- weight fp32 -> bf16 transpose -----------------------
__global__ void __launch_bounds__(256)
transpose_cvt(const float* __restrict__ W, bf16* __restrict__ Wt,
              int K, int N) {
    __shared__ bf16 tile[32][33];
    const int kb = blockIdx.y * 32, nb = blockIdx.x * 32;
    const int tx = threadIdx.x, ty = threadIdx.y;  // 32 x 8
#pragma unroll
    for (int r = 0; r < 32; r += 8)
        tile[ty + r][tx] = (bf16)W[(size_t)(kb + ty + r) * N + nb + tx];
    __syncthreads();
#pragma unroll
    for (int r = 0; r < 32; r += 8)
        Wt[(size_t)(nb + ty + r) * K + kb + tx] = tile[tx][ty + r];
}

// ------------------------------ LayerNorm ----------------------------------
__global__ void __launch_bounds__(256)
layernorm_bf16(const float* __restrict__ X, const float* __restrict__ gamma,
               const float* __restrict__ beta, bf16* __restrict__ Y) {
    const int row = blockIdx.x;
    const int t = threadIdx.x;
    const float4 v = ((const float4*)(X + (size_t)row * EMB))[t];

    __shared__ float sh[8];
    const int w = t >> 5, ln = t & 31;

    float s = v.x + v.y + v.z + v.w;
#pragma unroll
    for (int o = 16; o; o >>= 1) s += __shfl_xor(s, o, 32);
    if (ln == 0) sh[w] = s;
    __syncthreads();
    float mean = 0.f;
#pragma unroll
    for (int i = 0; i < 8; ++i) mean += sh[i];
    mean *= (1.0f / EMB);

    const float dx = v.x - mean, dy = v.y - mean, dz = v.z - mean,
                dw = v.w - mean;
    float s2 = dx * dx + dy * dy + dz * dz + dw * dw;
#pragma unroll
    for (int o = 16; o; o >>= 1) s2 += __shfl_xor(s2, o, 32);
    __syncthreads();
    if (ln == 0) sh[w] = s2;
    __syncthreads();
    float var = 0.f;
#pragma unroll
    for (int i = 0; i < 8; ++i) var += sh[i];
    var *= (1.0f / EMB);
    const float rstd = rsqrtf(var + 1e-5f);

    const float4 g = ((const float4*)gamma)[t];
    const float4 b = ((const float4*)beta)[t];
    bf16x4 o4;
    o4[0] = (bf16)(g.x * dx * rstd + b.x);
    o4[1] = (bf16)(g.y * dy * rstd + b.y);
    o4[2] = (bf16)(g.z * dz * rstd + b.z);
    o4[3] = (bf16)(g.w * dw * rstd + b.w);
    ((bf16x4*)Y)[(size_t)row * (EMB / 4) + t] = o4;
}

// ------------------------------- GEMM --------------------------------------
// C[M,N] = A[M,K] (bf16, row-major) x Bt[N,K]^T (bf16).
// Block = 128 threads (4 waves); block tile = 256x64 (wave w owns rows
// m0+w*64..+63).  B tile (64 rows x 32 K, 4KB) is double-buffered in LDS,
// filled by async global->LDS copies shared by all 4 waves.
// MODE 0: scatter bf16 to q/k buf   [b,h,t,d]
// MODE 1: scatter bf16 to v^T buf   [b,h,d,t]
// MODE 2: f32 out = res + acc + bias[n]
// MODE 3: bf16 out = gelu(acc + bias[n])
template <int MODE>
__global__ void __launch_bounds__(128)
gemm_wmma(const bf16* __restrict__ A, const bf16* __restrict__ Bt,
          int M, int N, int K,
          const float* __restrict__ bias, const float* __restrict__ res,
          void* __restrict__ out) {
    const int tid  = threadIdx.x;
    const int lane = tid & 31;
    const int wave = tid >> 5;
    const int m0 = blockIdx.y * 256 + wave * 64;
    const int n0 = blockIdx.x * 64;

    __shared__ bf16 Blds[2][64 * 32];  // two 4KB B stages

    // Issue async copy of B tile (64 x 32 bf16) for K-offset k into stage buf.
    // 256 x 16B chunks; 128 threads issue 2 async b128 copies each.
    auto issue_b = [&](int buf, int k) {
#pragma unroll
        for (int u = 0; u < 2; ++u) {
            const int c = tid + u * 128;        // chunk id 0..255
            const int row = c >> 2;             // 0..63   (N within tile)
            const int cc = c & 3;               // 16B chunk within 64B row
            const bf16* g = Bt + (size_t)(n0 + row) * K + k + cc * 8;
            const unsigned l =
                (unsigned)(size_t)(&Blds[buf][row * 32 + cc * 8]);
            asm volatile("global_load_async_to_lds_b128 %0, %1, off"
                         :: "v"(l), "v"((unsigned long long)(size_t)g)
                         : "memory");
        }
    };

    f32x8 acc[4][4];
#pragma unroll
    for (int i = 0; i < 4; ++i)
#pragma unroll
        for (int j = 0; j < 4; ++j) acc[i][j] = fzero();

    const int S = K / 32;  // K stages
    issue_b(0, 0);

    for (int s = 0; s < S; ++s) {
        const int k = s * 32;
        if (s + 1 < S) {
            issue_b((s + 1) & 1, k + 32);
            // two stages in flight; async loads complete in order, so
            // asynccnt<=2 means the current stage has landed in LDS.
            asm volatile("s_wait_asynccnt 0x2" ::: "memory");
        } else {
            asm volatile("s_wait_asynccnt 0x0" ::: "memory");
        }
        __syncthreads();

        bf16x16 af[4];
#pragma unroll
        for (int i = 0; i < 4; ++i)
            af[i] = load_frag_a(A + (size_t)(m0 + i * 16) * K + k, K, lane);

        if (k + 32 < K)  // pull next A slab toward L2/L0
            __builtin_prefetch(A + (size_t)(m0 + lane * 2) * K + k + 32, 0, 1);

        const bf16* Bl = &Blds[s & 1][0];
        bf16x16 bfv[4];
#pragma unroll
        for (int j = 0; j < 4; ++j)
            bfv[j] = load_frag_b(Bl + (size_t)(j * 16) * 32, 32, lane);

#pragma unroll
        for (int i = 0; i < 4; ++i)
#pragma unroll
            for (int j = 0; j < 4; ++j)
                acc[i][j] = wmma_bf16(af[i], bfv[j], acc[i][j]);

        __syncthreads();  // all waves done with this stage before reuse
    }

    const int half = lane >> 4, lo = lane & 15;
#pragma unroll
    for (int i = 0; i < 4; ++i)
#pragma unroll
        for (int j = 0; j < 4; ++j)
#pragma unroll
            for (int e = 0; e < 8; ++e) {
                const int m = m0 + i * 16 + e + half * 8;
                const int n = n0 + j * 16 + lo;
                const float v = acc[i][j][e];
                if (MODE == 0) {  // q/k: [b,h,t,d]
                    const int b = m >> 11, t = m & (SEQ - 1);
                    const int hh = n >> 6, d = n & (HD - 1);
                    ((bf16*)out)[(((size_t)(b * HEADS + hh)) * SEQ + t) * HD + d] =
                        (bf16)v;
                } else if (MODE == 1) {  // v^T: [b,h,d,t]
                    const int b = m >> 11, t = m & (SEQ - 1);
                    const int hh = n >> 6, d = n & (HD - 1);
                    ((bf16*)out)[(((size_t)(b * HEADS + hh)) * HD + d) * SEQ + t] =
                        (bf16)v;
                } else if (MODE == 2) {  // bias + residual, f32
                    ((float*)out)[(size_t)m * N + n] =
                        res[(size_t)m * N + n] + v + bias[n];
                } else {  // bias + gelu, bf16
                    ((bf16*)out)[(size_t)m * N + n] = (bf16)gelu_tanh(v + bias[n]);
                }
            }
}

// --------------------------- Flash attention -------------------------------
// One wave per (bh, 64-row q tile).  Causal: kv tiles 0..qtile only.
__global__ void __launch_bounds__(32)
attn_kernel(const bf16* __restrict__ q, const bf16* __restrict__ k,
            const bf16* __restrict__ vt, bf16* __restrict__ ctx) {
    const int lane = threadIdx.x;
    const int qi = blockIdx.x;  // q tile   0..31
    const int bh = blockIdx.y;  // b*16+h   0..31
    const int half = lane >> 4, lo = lane & 15;

    const bf16* Q  = q  + (size_t)bh * SEQ * HD + (size_t)qi * 64 * HD;
    const bf16* Kb = k  + (size_t)bh * SEQ * HD;
    const bf16* Vt = vt + (size_t)bh * HD * SEQ;

    __shared__ bf16 Plds[64 * 64];

    bf16x16 aq[4][2];
#pragma unroll
    for (int i = 0; i < 4; ++i)
#pragma unroll
        for (int s = 0; s < 2; ++s)
            aq[i][s] = load_frag_a(Q + (size_t)(i * 16) * HD + s * 32, HD, lane);

    f32x8 acc[4][4];
    float rmax[4][8], rsum[4][8];
#pragma unroll
    for (int i = 0; i < 4; ++i) {
#pragma unroll
        for (int j = 0; j < 4; ++j) acc[i][j] = fzero();
#pragma unroll
        for (int e = 0; e < 8; ++e) {
            rmax[i][e] = -__builtin_inff();
            rsum[i][e] = 0.0f;
        }
    }

    const float sc = 0.125f;  // 1/sqrt(HD)

    for (int t = 0; t <= qi; ++t) {
        f32x8 s4[4][4];
#pragma unroll
        for (int i = 0; i < 4; ++i)
#pragma unroll
            for (int j = 0; j < 4; ++j) s4[i][j] = fzero();

        bf16x16 bk[4][2];
#pragma unroll
        for (int j = 0; j < 4; ++j)
#pragma unroll
            for (int s = 0; s < 2; ++s)
                bk[j][s] = load_frag_b(Kb + (size_t)(t * 64 + j * 16) * HD + s * 32,
                                       HD, lane);
#pragma unroll
        for (int i = 0; i < 4; ++i)
#pragma unroll
            for (int j = 0; j < 4; ++j)
#pragma unroll
                for (int s = 0; s < 2; ++s)
                    s4[i][j] = wmma_bf16(aq[i][s], bk[j][s], s4[i][j]);

#pragma unroll
        for (int i = 0; i < 4; ++i)
#pragma unroll
            for (int j = 0; j < 4; ++j)
#pragma unroll
                for (int e = 0; e < 8; ++e) {
                    float sv = s4[i][j][e] * sc;
                    if (t == qi) {
                        const int mr = i * 16 + e + half * 8;
                        const int nc = j * 16 + lo;
                        if (nc > mr) sv = -3.0e38f;
                    }
                    s4[i][j][e] = sv;
                }

#pragma unroll
        for (int i = 0; i < 4; ++i)
#pragma unroll
            for (int e = 0; e < 8; ++e) {
                float tm = s4[i][0][e];
#pragma unroll
                for (int j = 1; j < 4; ++j) tm = fmaxf(tm, s4[i][j][e]);
#pragma unroll
                for (int o = 8; o; o >>= 1) tm = fmaxf(tm, __shfl_xor(tm, o, 32));
                const float nm = fmaxf(rmax[i][e], tm);
                const float corr = __expf(rmax[i][e] - nm);
                rmax[i][e] = nm;
                float ps = 0.0f;
#pragma unroll
                for (int j = 0; j < 4; ++j) {
                    const float p = __expf(s4[i][j][e] - nm);
                    s4[i][j][e] = p;
                    ps += p;
                }
#pragma unroll
                for (int o = 8; o; o >>= 1) ps += __shfl_xor(ps, o, 32);
                rsum[i][e] = rsum[i][e] * corr + ps;
#pragma unroll
                for (int j = 0; j < 4; ++j) acc[i][j][e] *= corr;
            }

        // P: C-layout -> LDS (bf16) -> A-layout
#pragma unroll
        for (int i = 0; i < 4; ++i)
#pragma unroll
            for (int j = 0; j < 4; ++j)
#pragma unroll
                for (int e = 0; e < 8; ++e) {
                    const int mr = i * 16 + e + half * 8;
                    const int nc = j * 16 + lo;
                    Plds[mr * 64 + nc] = (bf16)s4[i][j][e];
                }
        __syncthreads();

        bf16x16 bv[4][2];
#pragma unroll
        for (int j = 0; j < 4; ++j)
#pragma unroll
            for (int s = 0; s < 2; ++s)
                bv[j][s] = load_frag_b(Vt + (size_t)(j * 16) * SEQ + t * 64 + s * 32,
                                       SEQ, lane);
#pragma unroll
        for (int i = 0; i < 4; ++i) {
            bf16x16 ap0 = load_frag_a(Plds + (size_t)(i * 16) * 64 + 0, 64, lane);
            bf16x16 ap1 = load_frag_a(Plds + (size_t)(i * 16) * 64 + 32, 64, lane);
#pragma unroll
            for (int j = 0; j < 4; ++j) {
                acc[i][j] = wmma_bf16(ap0, bv[j][0], acc[i][j]);
                acc[i][j] = wmma_bf16(ap1, bv[j][1], acc[i][j]);
            }
        }
        __syncthreads();
    }

    const int b = bh >> 4, h = bh & 15;
#pragma unroll
    for (int i = 0; i < 4; ++i)
#pragma unroll
        for (int e = 0; e < 8; ++e) {
            const float inv = 1.0f / rsum[i][e];
            const int tok = b * SEQ + qi * 64 + i * 16 + e + half * 8;
#pragma unroll
            for (int j = 0; j < 4; ++j) {
                const int d = j * 16 + lo;
                ctx[(size_t)tok * EMB + h * HD + d] = (bf16)(acc[i][j][e] * inv);
            }
        }
}

// ------------------------------- launch ------------------------------------

extern "C" void kernel_launch(void* const* d_in, const int* in_sizes, int n_in,
                              void* d_out, int out_size, void* d_ws,
                              size_t ws_size, hipStream_t stream) {
    const float* x     = (const float*)d_in[0];
    const float* w_q   = (const float*)d_in[1];
    const float* w_k   = (const float*)d_in[2];
    const float* w_v   = (const float*)d_in[3];
    const float* w_o   = (const float*)d_in[4];
    const float* b_o   = (const float*)d_in[5];
    const float* ln1_s = (const float*)d_in[6];
    const float* ln1_b = (const float*)d_in[7];
    const float* ln2_s = (const float*)d_in[8];
    const float* ln2_b = (const float*)d_in[9];
    const float* w_ff1 = (const float*)d_in[10];
    const float* b_ff1 = (const float*)d_in[11];
    const float* w_ff2 = (const float*)d_in[12];
    const float* b_ff2 = (const float*)d_in[13];

    char* ws = (char*)d_ws;
    size_t off = 0;
    auto alloc = [&](size_t bytes) -> void* {
        void* p = ws + off;
        off += (bytes + 255) & ~(size_t)255;
        return p;
    };

    bf16* wqT   = (bf16*)alloc((size_t)EMB * EMB * 2);
    bf16* wkT   = (bf16*)alloc((size_t)EMB * EMB * 2);
    bf16* wvT   = (bf16*)alloc((size_t)EMB * EMB * 2);
    bf16* woT   = (bf16*)alloc((size_t)EMB * EMB * 2);
    bf16* wff1T = (bf16*)alloc((size_t)FF * EMB * 2);   // [FF, EMB]
    bf16* wff2T = (bf16*)alloc((size_t)EMB * FF * 2);   // [EMB, FF]
    bf16* h1    = (bf16*)alloc((size_t)TOK * EMB * 2);  // LN1(x)
    bf16* qbuf  = (bf16*)alloc((size_t)TOK * EMB * 2);  // [b,h,t,d]
    bf16* kbuf  = (bf16*)alloc((size_t)TOK * EMB * 2);  // [b,h,t,d]
    bf16* vtbuf = (bf16*)alloc((size_t)TOK * EMB * 2);  // [b,h,d,t]
    bf16* ctx   = (bf16*)alloc((size_t)TOK * EMB * 2);  // [token, EMB]
    float* x1   = (float*)alloc((size_t)TOK * EMB * 4); // x + attn
    bf16* h2    = (bf16*)alloc((size_t)TOK * EMB * 2);  // LN2(x1)
    bf16* ffa   = (bf16*)alloc((size_t)TOK * FF * 2);   // gelu(ff1)

    const dim3 tb(32, 8);
    transpose_cvt<<<dim3(EMB / 32, EMB / 32), tb, 0, stream>>>(w_q, wqT, EMB, EMB);
    transpose_cvt<<<dim3(EMB / 32, EMB / 32), tb, 0, stream>>>(w_k, wkT, EMB, EMB);
    transpose_cvt<<<dim3(EMB / 32, EMB / 32), tb, 0, stream>>>(w_v, wvT, EMB, EMB);
    transpose_cvt<<<dim3(EMB / 32, EMB / 32), tb, 0, stream>>>(w_o, woT, EMB, EMB);
    transpose_cvt<<<dim3(FF / 32, EMB / 32), tb, 0, stream>>>(w_ff1, wff1T, EMB, FF);
    transpose_cvt<<<dim3(EMB / 32, FF / 32), tb, 0, stream>>>(w_ff2, wff2T, FF, EMB);

    layernorm_bf16<<<TOK, 256, 0, stream>>>(x, ln1_s, ln1_b, h1);

    // QKV projections (M=4096, N=1024, K=1024); 256x64 block tiles
    gemm_wmma<0><<<dim3(EMB / 64, TOK / 256), 128, 0, stream>>>(
        h1, wqT, TOK, EMB, EMB, nullptr, nullptr, qbuf);
    gemm_wmma<0><<<dim3(EMB / 64, TOK / 256), 128, 0, stream>>>(
        h1, wkT, TOK, EMB, EMB, nullptr, nullptr, kbuf);
    gemm_wmma<1><<<dim3(EMB / 64, TOK / 256), 128, 0, stream>>>(
        h1, wvT, TOK, EMB, EMB, nullptr, nullptr, vtbuf);

    // Flash attention: 32 q-tiles x 32 (batch*head)
    attn_kernel<<<dim3(SEQ / 64, BATCH * HEADS), 32, 0, stream>>>(qbuf, kbuf,
                                                                  vtbuf, ctx);

    // Output projection + bias + residual -> x1 (f32)
    gemm_wmma<2><<<dim3(EMB / 64, TOK / 256), 128, 0, stream>>>(
        ctx, woT, TOK, EMB, EMB, b_o, x, x1);

    layernorm_bf16<<<TOK, 256, 0, stream>>>(x1, ln2_s, ln2_b, h2);

    // FF1 + bias + GELU (M=4096, N=4096, K=1024)
    gemm_wmma<3><<<dim3(FF / 64, TOK / 256), 128, 0, stream>>>(
        h2, wff1T, TOK, FF, EMB, b_ff1, nullptr, ffa);

    // FF2 + bias + residual -> d_out (f32)  (M=4096, N=1024, K=4096)
    gemm_wmma<2><<<dim3(EMB / 64, TOK / 256), 128, 0, stream>>>(
        ffa, wff2T, TOK, EMB, FF, b_ff2, x1, (float*)d_out);
}
